// MiniQwen3NextExperts_74517682586451
// MI455X (gfx1250) — matrix operand order repaired and checked by reading, not compiled, not run on hardware
//
#include <hip/hip_runtime.h>
#include <hip/hip_bf16.h>

#define TOKENS 4096
#define HIDDEN 2048
#define INTER  1024
#define NEXP   8
#define TOPK   2
#define TK     (TOKENS*TOPK)   /* 8192 routed slots */
#define CAP    TK              /* per-expert list capacity (worst case) */
#define MAX_TILES (TK/16)      /* 512 */

typedef __attribute__((ext_vector_type(16))) __bf16 v16bf;
typedef __attribute__((ext_vector_type(8)))  __bf16 v8bf;
typedef __attribute__((ext_vector_type(4)))  __bf16 v4bf;
typedef __attribute__((ext_vector_type(8)))  float  v8f;

// ---------------- fp32 -> bf16 conversion (4 elems / thread) ----------------
__global__ __launch_bounds__(256) void cvt_bf16(const float* __restrict__ s,
                                                __bf16* __restrict__ d, int n) {
    int i = (blockIdx.x * 256 + threadIdx.x) * 4;
    if (i + 3 < n) {
        float4 f = *(const float4*)(s + i);
        v4bf o = { (__bf16)f.x, (__bf16)f.y, (__bf16)f.z, (__bf16)f.w };
        *(v4bf*)(d + i) = o;
    }
}

// ---------------- routing: one wave32 per expert, stable order --------------
__global__ __launch_bounds__(32) void route_kernel(const int* __restrict__ idx,
                                                   int* __restrict__ lists,
                                                   int* __restrict__ counts) {
    int e = blockIdx.x;
    int lane = threadIdx.x;
    int count = 0;
    for (int base = 0; base < TK; base += 32) {
        int i = base + lane;                 // slot id = t*TOPK + k
        bool m = (idx[i] == e);
        unsigned long long mask = __ballot(m);
        int pre = __popcll(mask & ((1ull << lane) - 1ull));
        if (m) lists[e * CAP + count + pre] = i;
        count += __popcll(mask);
    }
    if (lane == 0) counts[e] = count;
}

// ------------- GEMM1: act = silu(X Gg^T) * (X Gu^T), bf16 WMMA --------------
// grid.x = NEXP*MAX_TILES (expert-major), grid.y = INTER/64, 128 thr = 4 waves
__global__ __launch_bounds__(128) void gemm_gateup(const __bf16* __restrict__ X,
                                                   const __bf16* __restrict__ G,
                                                   const int* __restrict__ lists,
                                                   const int* __restrict__ counts,
                                                   __bf16* __restrict__ act) {
    int e    = blockIdx.x / MAX_TILES;
    int tile = blockIdx.x % MAX_TILES;
    int cnt  = counts[e];
    if (tile * 16 >= cnt) return;

    int wave = threadIdx.x >> 5;
    int lane = threadIdx.x & 31;
    int hh = lane >> 4;          // half-wave
    int ln = lane & 15;

    __shared__ int slots[16];
    if (threadIdx.x < 16) {
        int ei = tile * 16 + threadIdx.x;
        slots[threadIdx.x] = (ei < cnt) ? lists[e * CAP + ei] : -1;
    }
    __syncthreads();

    int srow  = slots[ln];
    int token = (srow >= 0 ? srow : slots[0]) >> 1;   // slot -> token

    const __bf16* xrow  = X + (size_t)token * HIDDEN;
    int ocol = blockIdx.y * 64 + wave * 16 + ln;      // gate column
    const __bf16* grow = G + ((size_t)e * (2 * INTER) + ocol) * HIDDEN;
    const __bf16* urow = grow + (size_t)INTER * HIDDEN;

    v8f accG = {}; v8f accU = {};
    for (int k0 = 0; k0 < HIDDEN; k0 += 32) {
        union { v16bf v; v8bf h[2]; } a;
        a.h[0] = *(const v8bf*)(xrow + k0 + 16 * hh);       // K = 16h + j
        a.h[1] = *(const v8bf*)(xrow + k0 + 16 * hh + 8);   // K = 16h + 8 + j
        v16bf bg = *(const v16bf*)(grow + k0 + 16 * hh);    // K = 16h + j
        v16bf bu = *(const v16bf*)(urow + k0 + 16 * hh);
        accG = __builtin_amdgcn_wmma_f32_16x16x32_bf16(false, a.v, false, bg,
                                                       (short)0, accG, false, false);
        accU = __builtin_amdgcn_wmma_f32_16x16x32_bf16(false, a.v, false, bu,
                                                       (short)0, accU, false, false);
    }
#pragma unroll
    for (int v = 0; v < 8; ++v) {
        int s = slots[8 * hh + v];          // row M = 8h + v of C tile
        if (s >= 0) {
            float g = accG[v];
            float u = accU[v];
            float r = (g / (1.0f + __expf(-g))) * u;   // SiLU(g) * u
            act[(size_t)s * INTER + ocol] = (__bf16)r;
        }
    }
}

// ------------- GEMM2: dout[slot] = act[slot] @ Dn^T (fp32 out) --------------
// grid.x = NEXP*MAX_TILES, grid.y = HIDDEN/64, 128 thr = 4 waves
__global__ __launch_bounds__(128) void gemm_down(const __bf16* __restrict__ act,
                                                 const __bf16* __restrict__ D,
                                                 const int* __restrict__ lists,
                                                 const int* __restrict__ counts,
                                                 float* __restrict__ dout) {
    int e    = blockIdx.x / MAX_TILES;
    int tile = blockIdx.x % MAX_TILES;
    int cnt  = counts[e];
    if (tile * 16 >= cnt) return;

    int wave = threadIdx.x >> 5;
    int lane = threadIdx.x & 31;
    int hh = lane >> 4;
    int ln = lane & 15;

    __shared__ int slots[16];
    if (threadIdx.x < 16) {
        int ei = tile * 16 + threadIdx.x;
        slots[threadIdx.x] = (ei < cnt) ? lists[e * CAP + ei] : -1;
    }
    __syncthreads();

    int srow = slots[ln];
    int aslot = (srow >= 0 ? srow : slots[0]);
    const __bf16* arow = act + (size_t)aslot * INTER;
    int ocol = blockIdx.y * 64 + wave * 16 + ln;      // hidden column
    const __bf16* drow = D + ((size_t)e * HIDDEN + ocol) * INTER;

    v8f acc = {};
    for (int k0 = 0; k0 < INTER; k0 += 32) {
        union { v16bf v; v8bf h[2]; } a;
        a.h[0] = *(const v8bf*)(arow + k0 + 16 * hh);
        a.h[1] = *(const v8bf*)(arow + k0 + 16 * hh + 8);
        v16bf b = *(const v16bf*)(drow + k0 + 16 * hh);
        acc = __builtin_amdgcn_wmma_f32_16x16x32_bf16(false, a.v, false, b,
                                                      (short)0, acc, false, false);
    }
#pragma unroll
    for (int v = 0; v < 8; ++v) {
        int s = slots[8 * hh + v];
        if (s >= 0) dout[(size_t)s * HIDDEN + ocol] = acc[v];
    }
}

// ------------- combine: out[t] = w0*dout[2t] + w1*dout[2t+1] ----------------
__global__ __launch_bounds__(256) void combine_kernel(const float* __restrict__ w,
                                                      const float* __restrict__ dout,
                                                      float* __restrict__ out) {
    int i = blockIdx.x * 256 + threadIdx.x;
    if (i >= TOKENS * HIDDEN) return;
    int t = i / HIDDEN;
    int h = i - t * HIDDEN;
    out[i] = w[2 * t]     * dout[(size_t)(2 * t)     * HIDDEN + h]
           + w[2 * t + 1] * dout[(size_t)(2 * t + 1) * HIDDEN + h];
}

extern "C" void kernel_launch(void* const* d_in, const int* in_sizes, int n_in,
                              void* d_out, int out_size, void* d_ws, size_t ws_size,
                              hipStream_t stream) {
    const float* hs  = (const float*)d_in[0];   // [T, H]
    const int*   idx = (const int*)d_in[1];     // [T, K]
    const float* wts = (const float*)d_in[2];   // [T, K]
    const float* gup = (const float*)d_in[3];   // [E, 2I, H]
    const float* dwn = (const float*)d_in[4];   // [E, H, I]
    float* out = (float*)d_out;

    char* ws = (char*)d_ws;
    size_t off = 0;
    __bf16* Xbf = (__bf16*)(ws + off); off += (size_t)TOKENS * HIDDEN * 2;
    __bf16* Gbf = (__bf16*)(ws + off); off += (size_t)NEXP * 2 * INTER * HIDDEN * 2;
    __bf16* Dbf = (__bf16*)(ws + off); off += (size_t)NEXP * HIDDEN * INTER * 2;
    __bf16* Abf = (__bf16*)(ws + off); off += (size_t)TK * INTER * 2;
    float*  dow = (float*)(ws + off);  off += (size_t)TK * HIDDEN * 4;
    int* lists  = (int*)(ws + off);    off += (size_t)NEXP * CAP * 4;
    int* counts = (int*)(ws + off);    off += 64;

    const int nX = TOKENS * HIDDEN;
    const int nG = NEXP * 2 * INTER * HIDDEN;
    const int nD = NEXP * HIDDEN * INTER;
    cvt_bf16<<<nX / 1024, 256, 0, stream>>>(hs,  Xbf, nX);
    cvt_bf16<<<nG / 1024, 256, 0, stream>>>(gup, Gbf, nG);
    cvt_bf16<<<nD / 1024, 256, 0, stream>>>(dwn, Dbf, nD);

    route_kernel<<<NEXP, 32, 0, stream>>>(idx, lists, counts);

    gemm_gateup<<<dim3(NEXP * MAX_TILES, INTER / 64), 128, 0, stream>>>(
        Xbf, Gbf, lists, counts, Abf);

    gemm_down<<<dim3(NEXP * MAX_TILES, HIDDEN / 64), 128, 0, stream>>>(
        Abf, Dbf, lists, counts, dow);

    combine_kernel<<<(TOKENS * HIDDEN) / 256, 256, 0, stream>>>(wts, dow, out);
}
